// LayerwisePathwayMLP_SingleLayer_33981781246566
// MI455X (gfx1250) — matrix operand order repaired and checked by reading, not compiled, not run on hardware
//
#include <hip/hip_runtime.h>
#include <hip/hip_bf16.h>
#include <math.h>
#include <stdint.h>

typedef __attribute__((ext_vector_type(16))) __bf16 v16bf;
typedef __attribute__((ext_vector_type(8)))  __bf16 v8bf;
typedef __attribute__((ext_vector_type(8)))  float  v8f;
typedef unsigned int u32x4 __attribute__((ext_vector_type(4)));
typedef int          s32x8 __attribute__((ext_vector_type(8)));
typedef int          s32x4 __attribute__((ext_vector_type(4)));

#define WG1_ELEMS (512*896)
#define R1P_ELEMS (16*896)
#define W2P_ELEMS (16*512)
#define R2P_ELEMS (16*512)
#define WBUF_ELEMS (R1P_ELEMS + W2P_ELEMS + R2P_ELEMS)   // 30720 ushorts = 61440 B

__device__ __forceinline__ unsigned short f2bf(float f) {
  __bf16 b = (__bf16)f;
  return __builtin_bit_cast(unsigned short, b);
}

__device__ __forceinline__ v8f wmma_bf16(v16bf a, v16bf b, v8f c) {
  // emits v_wmma_f32_16x16x32_bf16
  return __builtin_amdgcn_wmma_f32_16x16x32_bf16(false, a, false, b, (short)0, c, false, false);
}

// A operand (16x32 bf16, MxK) from row-major [row][k] storage.
// Lane L holds row M=L%16; L<16 -> K 0..15, L>=16 -> K 16..31 (contiguous).
__device__ __forceinline__ v16bf loadA(const unsigned short* base, int stride,
                                       int mbase, int kbase, int lane) {
  int row  = mbase + (lane & 15);
  int koff = kbase + ((lane & 16) ? 16 : 0);
  const v8bf* p = (const v8bf*)(base + row * stride + koff);
  union { v16bf v; v8bf h[2]; } u;
  u.h[0] = p[0];   // K koff .. koff+7
  u.h[1] = p[1];   // K koff+8 .. koff+15
  return u.v;
}

// B operand (32x16 bf16, KxN) from row-major [n][k] storage.
// Lane L holds column N=L%16; L<16 -> K 0..15, L>=16 -> K 16..31 (contiguous 32B).
__device__ __forceinline__ v16bf loadB(const unsigned short* base, int stride,
                                       int nbase, int kbase, int lane) {
  int col  = nbase + (lane & 15);
  int koff = kbase + ((lane & 16) ? 16 : 0);
  return *(const v16bf*)(base + col * stride + koff);
}

// One-shot 1-D Tensor Data Mover copy: global -> LDS.  n8 = payload in 8-byte units
// (must be <= 65535).  D# layout per CDNA5 ISA ch.8 (group0: count/lds/global/type,
// group1: data_size=8B, tensor_dim0 = tile_dim0 = n8, single row).
// This toolchain exposes the 6-arg builtin (extra 8-DWORD group, zero-filled).
__device__ __forceinline__ void tdm_copy_1d(const void* gsrc, unsigned lds_byte_addr,
                                            unsigned n8) {
  unsigned long long ga = (unsigned long long)(uintptr_t)gsrc;
  u32x4 g0;
  g0.x = 1u;                                            // count=1 (valid), user mode
  g0.y = lds_byte_addr;                                 // D#.lds_addr
  g0.z = (unsigned)(ga & 0xffffffffu);                  // global_addr[31:0]
  g0.w = (unsigned)((ga >> 32) & 0x1ffffffu) | (2u << 30); // global_addr[56:32], type=2
  s32x8 g1;
  g1[0] = (3 << 16);                 // workgroup_mask=0, data_size=3 (8B), no flags
  g1[1] = (int)((n8 & 0xffffu) << 16);  // atomic_barrier_addr=0, tensor_dim0[15:0]
  g1[2] = (int)((n8 >> 16) & 0xffffu) | (1 << 16); // tensor_dim0[31:16], tensor_dim1=1
  g1[3] = (int)(n8 << 16);           // tensor_dim1[31:16]=0, tile_dim0=n8
  g1[4] = 1;                         // tile_dim1=1, tile_dim2=0 (unused)
  g1[5] = (int)n8;                   // tensor_dim0_stride[31:0]
  g1[6] = 0;                         // stride0[47:32]=0, tensor_dim1_stride[15:0]=0
  g1[7] = 0;
  s32x4 g2 = {1, 0, 0, 0};           // tensor_dim2=1, tile_dim3=0 (unused)
  s32x4 g3 = {0, 0, 0, 0};
  s32x8 g4 = {0, 0, 0, 0, 0, 0, 0, 0};
  __builtin_amdgcn_tensor_load_to_lds(g0, g1, g2, g3, g4, 0);
}

__device__ __forceinline__ void softmax16(float* p, const float* __restrict__ bias) {
  float v[16];
  float mx = -3.402823466e38f;
  #pragma unroll
  for (int j = 0; j < 16; ++j) { v[j] = p[j] + bias[j]; mx = fmaxf(mx, v[j]); }
  float s = 0.f;
  #pragma unroll
  for (int j = 0; j < 16; ++j) { v[j] = expf(v[j] - mx); s += v[j]; }
  float inv = 1.f / s;
  #pragma unroll
  for (int j = 0; j < 16; ++j) p[j] = v[j] * inv;
}

// ---- pack weights into grouped, padded bf16 layouts in workspace ----
__global__ void prep_kernel(const float* __restrict__ fc1w, const float* __restrict__ r1w,
                            const float* __restrict__ fc2w, const float* __restrict__ r2w,
                            const int* __restrict__ ig1, unsigned short* __restrict__ ws) {
  int i = blockIdx.x * 256 + threadIdx.x;
  unsigned short* wg1 = ws;
  unsigned short* r1p = ws + WG1_ELEMS;
  unsigned short* w2p = r1p + R1P_ELEMS;
  unsigned short* r2p = w2p + W2P_ELEMS;
  if (i < WG1_ELEMS) {
    int o = i / 896, t = i - o * 896, g = t / 224, j = t - g * 224;
    float v = (j < 196) ? fc1w[o * 784 + ig1[g * 196 + j]] : 0.f;
    wg1[i] = f2bf(v);
  } else if (i < WG1_ELEMS + R1P_ELEMS) {
    int k = i - WG1_ELEMS;
    int o = k / 896, t = k - o * 896, g = t / 224, j = t - g * 224;
    float v = (j < 196) ? r1w[o * 784 + ig1[g * 196 + j]] : 0.f;
    r1p[k] = f2bf(v);
  } else if (i < WG1_ELEMS + R1P_ELEMS + W2P_ELEMS) {
    int k = i - (WG1_ELEMS + R1P_ELEMS);
    int o = k / 512, c = k - o * 512;
    w2p[k] = f2bf((o < 8) ? fc2w[o * 512 + c] : 0.f);  // only outputs 0..7 are routed
  } else if (i < WG1_ELEMS + R1P_ELEMS + W2P_ELEMS + R2P_ELEMS) {
    int k = i - (WG1_ELEMS + R1P_ELEMS + W2P_ELEMS);
    r2p[k] = f2bf(r2w[k]);
  }
}

// ---- fully fused: routing1 -> pathway1 -> gelu -> routing2 -> pathway2 ----
__global__ void __launch_bounds__(256)
fused_kernel(const float* __restrict__ x,
             const float* __restrict__ fc1b, const float* __restrict__ fc2b,
             const float* __restrict__ r1b,  const float* __restrict__ r2b,
             const unsigned short* __restrict__ ws, float* __restrict__ out) {
  const unsigned short* wg1 = ws;

  __shared__ __align__(16) unsigned short xg[64 * 896];   // grouped/padded x, bf16 (112KB)
  __shared__ __align__(16) unsigned short hS[64 * 512];   // gelu(h) tile, bf16 (64KB)
  __shared__ __align__(16) unsigned short wbufS[WBUF_ELEMS]; // r1p|w2p|r2p via TDM (60KB)
  __shared__ float rw1[64 * 16];
  __shared__ float rw2[64 * 16];

  const unsigned short* r1pS = wbufS;
  const unsigned short* w2pS = wbufS + R1P_ELEMS;
  const unsigned short* r2pS = wbufS + R1P_ELEMS + W2P_ELEMS;

  const int tid  = threadIdx.x;
  const int wv   = tid >> 5;
  const int lane = tid & 31;
  const int hi8  = (lane & 16) ? 8 : 0;
  const int lcol = lane & 15;
  const size_t row0 = (size_t)blockIdx.x * 64;

  // ---- kick off TDM: routing + layer-2 weights -> LDS (overlaps x staging) ----
  if (wv == 0)
    tdm_copy_1d(ws + WG1_ELEMS, (unsigned)(uintptr_t)(const void*)wbufS,
                (WBUF_ELEMS * 2) / 8);

  // ---- stage x tile: fp32 global -> grouped/padded bf16 LDS ----
  const float* xr = x + row0 * 784;
  for (int i = tid; i < 64 * 784; i += 256) {
    int m = i / 784, p = i - m * 784;
    int rr = p / 28, cc = p - rr * 28;
    int g = ((rr >= 14) ? 2 : 0) + ((cc >= 14) ? 1 : 0);
    int j = (rr - ((rr >= 14) ? 14 : 0)) * 14 + (cc - ((cc >= 14) ? 14 : 0));
    xg[m * 896 + g * 224 + j] = f2bf(xr[(size_t)m * 784 + p]);
  }
  for (int i = tid; i < 64 * 4 * 28; i += 256) {  // zero the K padding (196..223)
    int m = i / 112, t2 = i - m * 112;
    int g = t2 / 28, j = 196 + (t2 - g * 28);
    xg[m * 896 + g * 224 + j] = 0;
  }
  if (wv == 0) __builtin_amdgcn_s_wait_tensorcnt(0);  // TDM writes visible pre-barrier
  __syncthreads();

  // ---- routing 1 logits via WMMA (waves 0..3, one 16-row tile each) ----
  if (wv < 4) {
    int mbase = wv * 16;
    v8f acc = {};
    #pragma unroll
    for (int kk = 0; kk < 28; ++kk)
      acc = wmma_bf16(loadA(xg, 896, mbase, kk * 32, lane),
                      loadB(r1pS, 896, 0, kk * 32, lane), acc);
    #pragma unroll
    for (int i = 0; i < 8; ++i)
      rw1[(mbase + i + hi8) * 16 + lcol] = acc[i];
  }
  __syncthreads();
  if (tid < 64) softmax16(rw1 + tid * 16, r1b);
  __syncthreads();

  // ---- layer 1: B tile loaded ONCE per (g,kk), 4 independent row-tile accumulators ----
  for (int t = wv; t < 32; t += 8) {       // 32 column tiles of 16 over the 512 outputs
    int obase = t * 16;
    int hgrp  = t >> 3;                    // output group h = o/128
    if (t + 8 < 32)                        // global_prefetch_b8 of next weight tile
      __builtin_prefetch(wg1 + (size_t)((t + 8) * 16 + lcol) * 896, 0, 1);
    float bcol = fc1b[obase + lcol];
    v8f tot[4] = {{}, {}, {}, {}};
    float ssum[4][8] = {};
    #pragma unroll
    for (int g = 0; g < 4; ++g) {
      v8f acc[4] = {{}, {}, {}, {}};
      #pragma unroll
      for (int kk = 0; kk < 7; ++kk) {
        int kb = g * 224 + kk * 32;
        v16bf bm = loadB(wg1, 896, obase, kb, lane);   // one global load per (g,kk)
        #pragma unroll
        for (int r = 0; r < 4; ++r)                    // 4 independent WMMA chains
          acc[r] = wmma_bf16(loadA(xg, 896, r * 16, kb, lane), bm, acc[r]);
      }
      #pragma unroll
      for (int r = 0; r < 4; ++r) {
        #pragma unroll
        for (int i = 0; i < 8; ++i) {    // C layout: reg i -> row mbase+i(+8 hi half)
          float s = rw1[(r * 16 + i + hi8) * 16 + g * 4 + hgrp];
          tot[r][i]  += s * acc[r][i];
          ssum[r][i] += s;
        }
      }
    }
    #pragma unroll
    for (int r = 0; r < 4; ++r) {
      #pragma unroll
      for (int i = 0; i < 8; ++i) {
        float v = tot[r][i] + bcol * ssum[r][i];
        v = 0.5f * v * (1.0f + erff(v * 0.70710678118654752f));  // exact GELU
        hS[(r * 16 + i + hi8) * 512 + obase + lcol] = f2bf(v);
      }
    }
  }
  __syncthreads();

  // ---- routing 2 logits via WMMA ----
  if (wv < 4) {
    int mbase = wv * 16;
    v8f acc = {};
    #pragma unroll
    for (int kk = 0; kk < 16; ++kk)
      acc = wmma_bf16(loadA(hS, 512, mbase, kk * 32, lane),
                      loadB(r2pS, 512, 0, kk * 32, lane), acc);
    #pragma unroll
    for (int i = 0; i < 8; ++i)
      rw2[(mbase + i + hi8) * 16 + lcol] = acc[i];
  }
  __syncthreads();
  if (tid < 64) softmax16(rw2 + tid * 16, r2b);
  __syncthreads();

  // ---- layer 2: contiguous 128-groups, routed combine, write (B,10) output ----
  if (wv < 4) {
    int mbase = wv * 16;
    v8f tot = {};
    float ssum[8] = {0.f, 0.f, 0.f, 0.f, 0.f, 0.f, 0.f, 0.f};
    int h2 = (lcol >> 1) & 3;              // out-group for column pair; cols>=8 discarded
    #pragma unroll
    for (int g = 0; g < 4; ++g) {
      v8f acc = {};
      #pragma unroll
      for (int kk = 0; kk < 4; ++kk) {
        int kb = g * 128 + kk * 32;
        acc = wmma_bf16(loadA(hS, 512, mbase, kb, lane),
                        loadB(w2pS, 512, 0, kb, lane), acc);
      }
      #pragma unroll
      for (int i = 0; i < 8; ++i) {
        float s = rw2[(mbase + i + hi8) * 16 + g * 4 + h2];
        tot[i]  += s * acc[i];
        ssum[i] += s;
      }
    }
    float bcol = (lcol < 10) ? fc2b[lcol] : 0.f;
    #pragma unroll
    for (int i = 0; i < 8; ++i) {
      size_t row = row0 + (size_t)(mbase + i + hi8);
      if (lcol < 8)       out[row * 10 + lcol] = tot[i] + bcol * ssum[i];
      else if (lcol < 10) out[row * 10 + lcol] = 0.f;   // unrouted outputs are exactly 0
    }
  }
}

extern "C" void kernel_launch(void* const* d_in, const int* in_sizes, int n_in,
                              void* d_out, int out_size, void* d_ws, size_t ws_size,
                              hipStream_t stream) {
  (void)in_sizes; (void)n_in; (void)out_size; (void)ws_size;
  const float* x    = (const float*)d_in[0];
  const float* fc1w = (const float*)d_in[1];
  const float* fc1b = (const float*)d_in[2];
  const float* fc2w = (const float*)d_in[3];
  const float* fc2b = (const float*)d_in[4];
  const float* r1w  = (const float*)d_in[5];
  const float* r1b  = (const float*)d_in[6];
  const float* r2w  = (const float*)d_in[7];
  const float* r2b  = (const float*)d_in[8];
  const int*   ig1  = (const int*)d_in[9];
  unsigned short* ws = (unsigned short*)d_ws;
  float* out = (float*)d_out;

  const int prep_total = WG1_ELEMS + R1P_ELEMS + W2P_ELEMS + R2P_ELEMS;
  hipLaunchKernelGGL(prep_kernel, dim3((prep_total + 255) / 256), dim3(256), 0, stream,
                     fc1w, r1w, fc2w, r2w, ig1, ws);

  hipLaunchKernelGGL(fused_kernel, dim3(65536 / 64), dim3(256), 0, stream,
                     x, fc1b, fc2b, r1b, r2b, ws, out);
}
